// sparse_bottleneck_3564822856158
// MI455X (gfx1250) — compile-verified, matrix operand order checked by reading, not run on hardware
//
#include <hip/hip_runtime.h>
#include <hip/hip_bf16.h>

#define N_SITES 100000
#define C_IN    256
#define C_MID   64
#define C_OUT   256
#define NTILES  (N_SITES / 16)   // 6250, exact
#define KTAPS   27

// swizzled-weight image layout in d_ws (elements of bf16):
//   [0, 216*512)        : W2^T fragments (216 KB)
//   [216*512, 248*512)  : W3^T fragments (32 KB)   -- contiguous with W2 image
//   [248*512, 280*512)  : W1^T fragments (32 KB)
//   [280*512, ...)      : h1 (N_SITES+1 rows x 64 bf16)
#define W2_FRAG_ELEMS (216 * 512)
#define W3_FRAG_ELEMS (32 * 512)
#define W1_FRAG_ELEMS (32 * 512)
#define H1_OFF        (280 * 512)

typedef __bf16 bf16_t;
typedef __attribute__((ext_vector_type(16))) __bf16 v16bf;
typedef __attribute__((ext_vector_type(8)))  __bf16 v8bf;
typedef __attribute__((ext_vector_type(8)))  float  v8f;
typedef __attribute__((ext_vector_type(4)))  float  v4f;
typedef __attribute__((ext_vector_type(4)))  unsigned int u32x4;
typedef __attribute__((ext_vector_type(8)))  int  i32x8;
typedef __attribute__((ext_vector_type(4)))  int  i32x4;

static __device__ __forceinline__ v8f wmma_bf16(v16bf a, v16bf b, v8f c) {
  return __builtin_amdgcn_wmma_f32_16x16x32_bf16(false, a, false, b, (short)0, c,
                                                 false, false);
}

// CDNA5 16-bit operand layout (16x32 A == 32x16 B, bit-identical):
//   lane L: idx = L%16 (M for A / N for B), group g = L/16
//   per lane: elems 0..7 = K = Kbase + 8g + 0..7, elems 8..15 = Kbase+16+8g+0..7
static __device__ __forceinline__ void kpos(int c32, int& g, int& q) {
  g = (c32 >> 3) & 1;
  int j = ((c32 & 16) ? 4 : 0) + ((c32 & 7) >> 1);
  q = 2 * j + (c32 & 1);
}

static __device__ __forceinline__ v16bf lds_frag(const bf16_t* p) {
  v8bf lo = *(const v8bf*)p;
  v8bf hi = *(const v8bf*)(p + 8);
  v16bf r;
#pragma unroll
  for (int i = 0; i < 8; ++i) { r[i] = lo[i]; r[8 + i] = hi[i]; }
  return r;
}

static __device__ __forceinline__ v16bf bfrag_bf16(const bf16_t* row, int s, int g) {
  v8bf lo = *(const v8bf*)(row + s * 32 + 8 * g);
  v8bf hi = *(const v8bf*)(row + s * 32 + 16 + 8 * g);
  v16bf r;
#pragma unroll
  for (int i = 0; i < 8; ++i) { r[i] = lo[i]; r[8 + i] = hi[i]; }
  return r;
}

static __device__ __forceinline__ v16bf bfrag_f32(const float* row, int s, int g) {
  v4f a0 = *(const v4f*)(row + s * 32 + 8 * g);
  v4f a1 = *(const v4f*)(row + s * 32 + 8 * g + 4);
  v4f a2 = *(const v4f*)(row + s * 32 + 16 + 8 * g);
  v4f a3 = *(const v4f*)(row + s * 32 + 16 + 8 * g + 4);
  v16bf r;
#pragma unroll
  for (int i = 0; i < 4; ++i) {
    r[i]      = (bf16_t)a0[i];
    r[4 + i]  = (bf16_t)a1[i];
    r[8 + i]  = (bf16_t)a2[i];
    r[12 + i] = (bf16_t)a3[i];
  }
  return r;
}

// ---- Tensor Data Mover: one contiguous row of n8 8-byte elements -> LDS.
// D# per CDNA5 ISA. Issued by one wave; EXEC ignored.
static __device__ __forceinline__ void tdm_load_to_lds(const void* gsrc,
                                                       void* lds_dst,
                                                       unsigned bytes) {
  unsigned long long ga = (unsigned long long)gsrc;
  unsigned ldsoff = (unsigned)(unsigned long long)lds_dst;  // low 32b = LDS offset
  unsigned n8 = bytes >> 3;                                 // 8-byte elements
  u32x4 g0;
  g0[0] = 1u;                                          // count=1 (valid user D#)
  g0[1] = ldsoff;                                      // lds_addr
  g0[2] = (unsigned)(ga & 0xFFFFFFFFull);              // global_addr[31:0]
  g0[3] = (unsigned)((ga >> 32) & 0x1FFFFFFull) | (2u << 30);  // addr[56:32]|type=2
  i32x8 g1;
  g1[0] = (int)(3u << 16);                 // workgroup_mask=0, data_size=8B
  g1[1] = (int)((n8 & 0xFFFFu) << 16);     // tensor_dim0[15:0]  @ bits 63:48
  g1[2] = (int)(((n8 >> 16) & 0xFFFFu) | (1u << 16));  // dim0[31:16] | dim1=1
  g1[3] = (int)((n8 & 0xFFFFu) << 16);     // tile_dim0 @ bits 127:112
  g1[4] = 1;                               // tile_dim1=1, tile_dim2=0 (2D)
  g1[5] = (int)n8;                         // tensor_dim0_stride[31:0]
  g1[6] = 0;
  g1[7] = 0;
  i32x4 z4 = {0, 0, 0, 0};
#if __clang_major__ >= 23
  i32x8 z8 = {0, 0, 0, 0, 0, 0, 0, 0};
  __builtin_amdgcn_tensor_load_to_lds(g0, g1, z4, z4, z8, 0);
#else
  __builtin_amdgcn_tensor_load_to_lds(g0, g1, z4, z4, 0);
#endif
}

// ---------- Kernel 0: build swizzled bf16 fragment images in workspace ----------
__global__ __launch_bounds__(256) void k0_swizzle(
    const float* __restrict__ W1, const float* __restrict__ W2,
    const float* __restrict__ W3, bf16_t* __restrict__ wsz)
{
  const int tid0 = blockIdx.x * blockDim.x + threadIdx.x;
  const int stride = gridDim.x * blockDim.x;
  for (int idx = tid0; idx < KTAPS * C_MID * C_MID; idx += stride) {
    int d = idx & 63, c = (idx >> 6) & 63, k = idx >> 12;   // W2[k][c][d]
    int t = d >> 4, s = c >> 5;
    int g, q; kpos(c & 31, g, q);
    wsz[(k * 8 + t * 2 + s) * 512 + (g * 16 + (d & 15)) * 16 + q] = (bf16_t)W2[idx];
  }
  for (int idx = tid0; idx < C_MID * C_OUT; idx += stride) {
    int d = idx & 255, c = idx >> 8;                        // W3[c][d]
    int u = d >> 4, s = c >> 5;
    int g, q; kpos(c & 31, g, q);
    wsz[W2_FRAG_ELEMS + (u * 2 + s) * 512 + (g * 16 + (d & 15)) * 16 + q] =
        (bf16_t)W3[idx];
  }
  for (int idx = tid0; idx < C_IN * C_MID; idx += stride) {
    int d = idx & 63, c = idx >> 6;                         // W1[c][d]
    int t = d >> 4, s = c >> 5;
    int g, q; kpos(c & 31, g, q);
    wsz[W2_FRAG_ELEMS + W3_FRAG_ELEMS + (t * 8 + s) * 512 +
        (g * 16 + (d & 15)) * 16 + q] = (bf16_t)W1[idx];
  }
}

// ---------- Kernel 1: h1^T = W1^T x X^T ; h1 = relu(g1*(X@W1)+b1) as bf16 ----------
__global__ __launch_bounds__(256) void k1_conv1(
    const float* __restrict__ X, const bf16_t* __restrict__ w1frag,
    const float* __restrict__ g1, const float* __restrict__ b1,
    bf16_t* __restrict__ h1)
{
  __shared__ bf16_t sW[W1_FRAG_ELEMS];  // 32 KB
  if (threadIdx.x < 32) {               // wave 0: one TDM DMA for whole image
    tdm_load_to_lds(w1frag, sW, W1_FRAG_ELEMS * (unsigned)sizeof(bf16_t));
    __builtin_amdgcn_s_wait_tensorcnt(0);
  }
  __syncthreads();

  const int tid = threadIdx.x;
  const int lane = tid & 31, wave = tid >> 5;
  const int g = lane >> 4, m = lane & 15;
  const int nw = (blockDim.x >> 5) * gridDim.x;
  int wid = blockIdx.x * (blockDim.x >> 5) + wave;

  // zero the pad row (neighbor index == N_SITES reads zeros)
  if (blockIdx.x == 0 && wave == 0)
    ((unsigned int*)(h1 + (size_t)N_SITES * C_MID))[lane] = 0u;

  v8f g1v[4], b1v[4];  // D layout: lane holds channel = 16t + 8g + r, site = m
#pragma unroll
  for (int t = 0; t < 4; ++t)
#pragma unroll
    for (int r = 0; r < 8; ++r) {
      g1v[t][r] = g1[16 * t + 8 * g + r];
      b1v[t][r] = b1[16 * t + 8 * g + r];
    }

  for (int tile = wid; tile < NTILES; tile += nw) {
    const float* rowp = X + (size_t)(tile * 16 + m) * C_IN;
    v8f acc[4] = {};
#pragma unroll
    for (int s = 0; s < 8; ++s) {          // K = 256 in 8 steps of 32
      v16bf b = bfrag_f32(rowp, s, g);
#pragma unroll
      for (int t = 0; t < 4; ++t) {
        v16bf a = lds_frag(sW + (t * 8 + s) * 512 + lane * 16);
        acc[t] = wmma_bf16(a, b, acc[t]);
      }
    }
    bf16_t* op = h1 + (size_t)(tile * 16 + m) * C_MID;
#pragma unroll
    for (int t = 0; t < 4; ++t) {
      v8bf o;
#pragma unroll
      for (int r = 0; r < 8; ++r) {
        float v = acc[t][r] * g1v[t][r] + b1v[t][r];
        o[r] = (bf16_t)fmaxf(v, 0.0f);
      }
      *(v8bf*)(op + 16 * t + 8 * g) = o;
    }
  }
}

// ---- Kernel 2: fused 27-tap gather-GEMM + affine/relu + conv3 + residual ----
__global__ __launch_bounds__(256) void k2_conv2_conv3(
    const bf16_t* __restrict__ h1, const bf16_t* __restrict__ wfrag,
    const float* __restrict__ g2, const float* __restrict__ b2,
    const float* __restrict__ g3, const float* __restrict__ b3,
    const float* __restrict__ X, const int* __restrict__ nb,
    float* __restrict__ out)
{
  extern __shared__ char smem_raw[];            // 248 KB: W2^T + W3^T frag image
  bf16_t* sW2 = (bf16_t*)smem_raw;              // frag (k*8 + t*2 + s)
  bf16_t* sW3 = (bf16_t*)smem_raw + W2_FRAG_ELEMS;  // frag (u*2 + s)

  if (threadIdx.x < 32) {  // wave 0: single 248 KB TDM DMA
    tdm_load_to_lds(wfrag, sW2,
                    (W2_FRAG_ELEMS + W3_FRAG_ELEMS) * (unsigned)sizeof(bf16_t));
    __builtin_amdgcn_s_wait_tensorcnt(0);
  }
  __syncthreads();

  const int tid = threadIdx.x;
  const int lane = tid & 31, wave = tid >> 5;
  const int g = lane >> 4, m = lane & 15;
  const int nw = (blockDim.x >> 5) * gridDim.x;
  int wid = blockIdx.x * (blockDim.x >> 5) + wave;

  v8f g2v[4], b2v[4];  // h2^T layout: lane holds channel = 16t + 8g + r
#pragma unroll
  for (int t = 0; t < 4; ++t)
#pragma unroll
    for (int r = 0; r < 8; ++r) {
      g2v[t][r] = g2[16 * t + 8 * g + r];
      b2v[t][r] = b2[16 * t + 8 * g + r];
    }

  for (int tile = wid; tile < NTILES; tile += nw) {
    const int site = tile * 16 + m;
    const int* nbrow = nb + (size_t)site * KTAPS;
    v8f acc2[4] = {};
    int ridx = nbrow[0];
    for (int k = 0; k < KTAPS; ++k) {
      int ridx_next = (k + 1 < KTAPS) ? nbrow[k + 1] : 0;  // prefetch index
      const bf16_t* rp = h1 + (size_t)ridx * C_MID;        // pad row => zeros
#pragma unroll
      for (int s = 0; s < 2; ++s) {                        // K = 64 in 2 steps
        v16bf b = bfrag_bf16(rp, s, g);
#pragma unroll
        for (int t = 0; t < 4; ++t) {
          v16bf a = lds_frag(sW2 + (k * 8 + t * 2 + s) * 512 + lane * 16);
          acc2[t] = wmma_bf16(a, b, acc2[t]);
        }
      }
      ridx = ridx_next;
    }
    // affine+relu; repack h2^T accumulators directly into conv3 B-fragments
    v16bf hb[2];
#pragma unroll
    for (int t = 0; t < 4; ++t)
#pragma unroll
      for (int r = 0; r < 8; ++r) {
        float v = fmaxf(acc2[t][r] * g2v[t][r] + b2v[t][r], 0.0f);
        hb[t >> 1][(t & 1) * 8 + r] = (bf16_t)v;
      }

    const float* xrow = X + (size_t)site * C_OUT;
    float* orow = out + (size_t)site * C_OUT;
#pragma unroll
    for (int u = 0; u < 16; ++u) {      // 256 output channels in 16 tiles
      v16bf a0 = lds_frag(sW3 + (u * 2 + 0) * 512 + lane * 16);
      v16bf a1 = lds_frag(sW3 + (u * 2 + 1) * 512 + lane * 16);
      v8f acc3 = {};
      acc3 = wmma_bf16(a0, hb[0], acc3);
      acc3 = wmma_bf16(a1, hb[1], acc3);
      const int ch0 = 16 * u + 8 * g;   // lane holds channels ch0..ch0+7
      v4f gA = *(const v4f*)(g3 + ch0), gB = *(const v4f*)(g3 + ch0 + 4);
      v4f bA = *(const v4f*)(b3 + ch0), bB = *(const v4f*)(b3 + ch0 + 4);
      v4f x0 = *(const v4f*)(xrow + ch0), x1 = *(const v4f*)(xrow + ch0 + 4);
      v4f o0, o1;
#pragma unroll
      for (int r = 0; r < 4; ++r) {
        o0[r] = fmaxf(acc3[r] * gA[r] + bA[r] + x0[r], 0.0f);
        o1[r] = fmaxf(acc3[4 + r] * gB[r] + bB[r] + x1[r], 0.0f);
      }
      *(v4f*)(orow + ch0) = o0;
      *(v4f*)(orow + ch0 + 4) = o1;
    }
  }
}

extern "C" void kernel_launch(void* const* d_in, const int* in_sizes, int n_in,
                              void* d_out, int out_size, void* d_ws, size_t ws_size,
                              hipStream_t stream) {
  (void)in_sizes; (void)n_in; (void)out_size; (void)ws_size;
  const float* X  = (const float*)d_in[0];
  const float* W1 = (const float*)d_in[1];
  const float* W2 = (const float*)d_in[2];
  const float* W3 = (const float*)d_in[3];
  const float* g1 = (const float*)d_in[4];
  const float* b1 = (const float*)d_in[5];
  const float* g2 = (const float*)d_in[6];
  const float* b2 = (const float*)d_in[7];
  const float* g3 = (const float*)d_in[8];
  const float* b3 = (const float*)d_in[9];
  const int*   nb = (const int*)d_in[10];

  bf16_t* wsz = (bf16_t*)d_ws;            // swizzled fragment images (280 KB)
  bf16_t* h1  = wsz + H1_OFF;             // (N_SITES+1) x 64 bf16 (~12.8 MB)

  k0_swizzle<<<128, 256, 0, stream>>>(W1, W2, W3, wsz);
  k1_conv1<<<256, 256, 0, stream>>>(X, wsz + W2_FRAG_ELEMS + W3_FRAG_ELEMS,
                                    g1, b1, h1);
  k2_conv2_conv3<<<256, 256,
                   (W2_FRAG_ELEMS + W3_FRAG_ELEMS) * (int)sizeof(bf16_t), stream>>>(
      h1, wsz, g2, b2, g3, b3, X, nb, (float*)d_out);
}
